// MethodNameLongformerEncoder_8632884265138
// MI455X (gfx1250) — matrix operand order repaired and checked by reading, not compiled, not run on hardware
//
#include <hip/hip_runtime.h>
#include <hip/hip_bf16.h>
#include <math.h>

// ---------------------------------------------------------------------------
// Longformer encoder forward (6 layers) for MI455X / gfx1250.
// All matmuls (QKV/O projections, FFN, QK^T, PV) run on v_wmma_f32_16x16x32_bf16
// (bf16 inputs, f32 accumulate). WMMA fragments are assembled strictly in
// registers: uint4 loads -> ext-vector literal -> __builtin_bit_cast. No
// address is ever taken of a fragment local, so no alloca / scratch traffic.
// ---------------------------------------------------------------------------

#define DMODEL 768
#define DFF    3072
#define NHEADS 12
#define DHEAD  64
#define SEQLEN 4096
#define NBATCH 2
#define NLAYER 6
#define CHUNK  256
#define NCHUNK (SEQLEN / CHUNK)     // 16
#define MROWS  (SEQLEN * NBATCH)    // 8192
#define PADTOK 1
#define NEGV   (-1000000000.0f)
#define LNEPS  1e-12f

typedef __attribute__((ext_vector_type(16))) __bf16       bf16x16;
typedef __attribute__((ext_vector_type(8)))  float        f32x8;
typedef __attribute__((ext_vector_type(8)))  unsigned int u32x8;

__device__ __forceinline__ f32x8 wmma_bf16(bf16x16 a, bf16x16 b, f32x8 c) {
  // 8 args: (neg_a, A, neg_b, B, c_mod, C, reuse_a, reuse_b)
  return __builtin_amdgcn_wmma_f32_16x16x32_bf16(false, a, false, b, (short)0, c,
                                                 false, false);
}

// Assemble a 16-elem bf16 fragment from two dword4 values, in registers only.
__device__ __forceinline__ bf16x16 frag_from(uint4 lo, uint4 hi) {
  u32x8 u = {lo.x, lo.y, lo.z, lo.w, hi.x, hi.y, hi.z, hi.w};
  return __builtin_bit_cast(bf16x16, u);
}

// Zero-select a dword4 without going through memory (lowers to v_cndmask).
__device__ __forceinline__ uint4 sel4(bool c, uint4 a) {
  uint4 r;
  r.x = c ? a.x : 0u;
  r.y = c ? a.y : 0u;
  r.z = c ? a.z : 0u;
  r.w = c ? a.w : 0u;
  return r;
}

// Load fragment: p0 -> elems 0..7, p1 -> elems 8..15 (16B-aligned pointers).
__device__ __forceinline__ bf16x16 load_frag2(const __bf16* p0, const __bf16* p1) {
  uint4 lo = *reinterpret_cast<const uint4*>(p0);
  uint4 hi = *reinterpret_cast<const uint4*>(p1);
  return frag_from(lo, hi);
}

__device__ __forceinline__ float gelu_tanh(float x) {
  const float k0 = 0.7978845608028654f; // sqrt(2/pi)
  return 0.5f * x * (1.0f + tanhf(k0 * (x + 0.044715f * x * x * x)));
}

__device__ __forceinline__ float wave_sum32(float x) {
#pragma unroll
  for (int off = 16; off > 0; off >>= 1) x += __shfl_xor(x, off, 32);
  return x;
}
__device__ __forceinline__ float half_max16(float x) {
#pragma unroll
  for (int off = 1; off < 16; off <<= 1) x = fmaxf(x, __shfl_xor(x, off, 32));
  return x;
}
__device__ __forceinline__ float half_sum16(float x) {
#pragma unroll
  for (int off = 1; off < 16; off <<= 1) x += __shfl_xor(x, off, 32);
  return x;
}

// ---------------------------------------------------------------------------
// WMMA GEMM: C[M,N] = A[M,K](bf16) * Wt[N,K]^T(bf16) + bias, optional GELU.
// Block = 256 threads = 8 waves; wave computes a 16x64 strip; block 128x64.
// ---------------------------------------------------------------------------
template <int ACT, bool WF32, bool WBF16>
__global__ __launch_bounds__(256) void gemm_wmma(const __bf16* __restrict__ A,
                                                 const __bf16* __restrict__ Wt,
                                                 const float* __restrict__ bias,
                                                 float* __restrict__ Cf,
                                                 __bf16* __restrict__ Cb,
                                                 int M, int N, int K) {
  const int lane = threadIdx.x & 31;
  const int wave = threadIdx.x >> 5;
  const int mr = lane & 15;
  const int hl = lane >> 4;
  const int m0 = blockIdx.y * 128 + wave * 16;
  const int n0 = blockIdx.x * 64;

  f32x8 acc[4];
#pragma unroll
  for (int t = 0; t < 4; ++t)
#pragma unroll
    for (int i = 0; i < 8; ++i) acc[t][i] = 0.0f;

  // A fragment: lane m = lane&15; elems: K = hl*16 + i
  const __bf16* Ap = A + (size_t)(m0 + mr) * K + hl * 16;
  // B fragment from Wt[N][K]: lane n = lane&15; elems: K = hl*16 + i
  const __bf16* Bp0 = Wt + (size_t)(n0 + mr) * K + hl * 16;

  for (int k = 0; k < K; k += 32) {
    __builtin_prefetch(Ap + k + 64, 0, 3);
    bf16x16 a = load_frag2(Ap + k, Ap + k + 8);
#pragma unroll
    for (int t = 0; t < 4; ++t) {
      const __bf16* bp = Bp0 + (size_t)t * 16 * K + k;
      bf16x16 b = load_frag2(bp, bp + 8);
      acc[t] = wmma_bf16(a, b, acc[t]);
    }
  }

  // C layout: VGPR r -> row m0 + hl*8 + r, col = n0 + t*16 + (lane&15)
  const int mh = hl * 8;
#pragma unroll
  for (int t = 0; t < 4; ++t) {
    const int col = n0 + t * 16 + mr;
    const float bv = bias[col];
#pragma unroll
    for (int r = 0; r < 8; ++r) {
      float v = acc[t][r] + bv;
      if (ACT == 1) v = gelu_tanh(v);
      const size_t idx = (size_t)(m0 + mh + r) * N + col;
      if (WF32) Cf[idx] = v;
      if (WBF16) Cb[idx] = (__bf16)v;
    }
  }
}

// W[K][N] f32 -> Wt[N][K] bf16
__global__ __launch_bounds__(256) void convert_wT(const float* __restrict__ W,
                                                  __bf16* __restrict__ Wt,
                                                  int K, int N) {
  int idx = blockIdx.x * 256 + threadIdx.x;
  if (idx >= K * N) return;
  int k = idx / N;
  int n = idx - k * N;
  Wt[(size_t)n * K + k] = (__bf16)W[idx];
}

// ---------------------------------------------------------------------------
// Embedding + LayerNorm. One block (256 thr) per (s, b) row of 768.
// ---------------------------------------------------------------------------
__global__ __launch_bounds__(256) void embed_ln(const int* __restrict__ tok,
                                                const float* __restrict__ wemb,
                                                const float* __restrict__ pemb,
                                                const float* __restrict__ gamma,
                                                const float* __restrict__ beta,
                                                float* __restrict__ h,
                                                __bf16* __restrict__ hb,
                                                int* __restrict__ attend) {
  __shared__ float red[8];
  __shared__ float stat;
  const int s = blockIdx.x;
  const int b = blockIdx.y;
  const int lane = threadIdx.x & 31;
  const int wave = threadIdx.x >> 5;
  const int t = tok[s * NBATCH + b];   // code_tokens[s, b]
  if (threadIdx.x == 0) attend[(size_t)b * SEQLEN + s] = (t != PADTOK) ? 1 : 0;

  const float* we = wemb + (size_t)t * DMODEL;
  const float* pe = pemb + (size_t)s * DMODEL;
  float v[3];
  float lsum = 0.0f;
#pragma unroll
  for (int i = 0; i < 3; ++i) {
    int d = threadIdx.x + i * 256;
    v[i] = we[d] + pe[d];
    lsum += v[i];
  }
  lsum = wave_sum32(lsum);
  if (lane == 0) red[wave] = lsum;
  __syncthreads();
  if (threadIdx.x == 0) {
    float a = 0.f;
#pragma unroll
    for (int i = 0; i < 8; ++i) a += red[i];
    stat = a * (1.0f / DMODEL);
  }
  __syncthreads();
  const float mean = stat;
  float lvar = 0.0f;
#pragma unroll
  for (int i = 0; i < 3; ++i) {
    float d0 = v[i] - mean;
    lvar += d0 * d0;
  }
  lvar = wave_sum32(lvar);
  __syncthreads();
  if (lane == 0) red[wave] = lvar;
  __syncthreads();
  if (threadIdx.x == 0) {
    float a = 0.f;
#pragma unroll
    for (int i = 0; i < 8; ++i) a += red[i];
    stat = a * (1.0f / DMODEL);
  }
  __syncthreads();
  const float inv = rsqrtf(stat + LNEPS);
  float* hr = h + ((size_t)b * SEQLEN + s) * DMODEL;
  __bf16* hbr = hb + ((size_t)b * SEQLEN + s) * DMODEL;
#pragma unroll
  for (int i = 0; i < 3; ++i) {
    int d = threadIdx.x + i * 256;
    float o = (v[i] - mean) * inv * gamma[d] + beta[d];
    hr[d] = o;
    hbr[d] = (__bf16)o;
  }
}

// h = LN(x + y); writes f32 + bf16. hout may alias x (row-private).
__global__ __launch_bounds__(256) void add_ln(const float* x, const float* y,
                                              const float* __restrict__ gamma,
                                              const float* __restrict__ beta,
                                              float* hout,
                                              __bf16* __restrict__ hbout) {
  __shared__ float red[8];
  __shared__ float stat;
  const size_t row = blockIdx.x;
  const int lane = threadIdx.x & 31;
  const int wave = threadIdx.x >> 5;
  const float* xr = x + row * DMODEL;
  const float* yr = y + row * DMODEL;
  float v[3];
  float lsum = 0.0f;
#pragma unroll
  for (int i = 0; i < 3; ++i) {
    int d = threadIdx.x + i * 256;
    v[i] = xr[d] + yr[d];
    lsum += v[i];
  }
  lsum = wave_sum32(lsum);
  if (lane == 0) red[wave] = lsum;
  __syncthreads();
  if (threadIdx.x == 0) {
    float a = 0.f;
#pragma unroll
    for (int i = 0; i < 8; ++i) a += red[i];
    stat = a * (1.0f / DMODEL);
  }
  __syncthreads();
  const float mean = stat;
  float lvar = 0.0f;
#pragma unroll
  for (int i = 0; i < 3; ++i) {
    float d0 = v[i] - mean;
    lvar += d0 * d0;
  }
  lvar = wave_sum32(lvar);
  __syncthreads();
  if (lane == 0) red[wave] = lvar;
  __syncthreads();
  if (threadIdx.x == 0) {
    float a = 0.f;
#pragma unroll
    for (int i = 0; i < 8; ++i) a += red[i];
    stat = a * (1.0f / DMODEL);
  }
  __syncthreads();
  const float inv = rsqrtf(stat + LNEPS);
  float* hr = hout + row * DMODEL;
  __bf16* hbr = hbout + row * DMODEL;
#pragma unroll
  for (int i = 0; i < 3; ++i) {
    int d = threadIdx.x + i * 256;
    float o = (v[i] - mean) * inv * gamma[d] + beta[d];
    hr[d] = o;
    hbr[d] = (__bf16)o;
  }
}

// Q,K,V f32 [B*S, 768] -> Qh/Kh bf16 [B,H,S,64], Vt bf16 [B,H,64,S]
__global__ __launch_bounds__(256) void pack_qkv(const float* __restrict__ q,
                                                const float* __restrict__ k,
                                                const float* __restrict__ v,
                                                __bf16* __restrict__ qh,
                                                __bf16* __restrict__ kh,
                                                __bf16* __restrict__ vt) {
  size_t idx = (size_t)blockIdx.x * 256 + threadIdx.x;
  if (idx >= (size_t)MROWS * DMODEL) return;
  int row = (int)(idx / DMODEL);
  int col = (int)(idx - (size_t)row * DMODEL);
  int b = row / SEQLEN;
  int s = row - b * SEQLEN;
  int hd = col >> 6;
  int d = col & 63;
  size_t bh = (size_t)b * NHEADS + hd;
  qh[(bh * SEQLEN + s) * DHEAD + d] = (__bf16)q[idx];
  kh[(bh * SEQLEN + s) * DHEAD + d] = (__bf16)k[idx];
  vt[(bh * DHEAD + d) * SEQLEN + s] = (__bf16)v[idx];
}

// ---------------------------------------------------------------------------
// Sliding-window attention, flash-style. Block = (chunk, head, batch),
// 512 threads = 16 waves, ONE 16-query tile per wave. Streams the 768-key
// window in 24 blocks of 32 keys. Fragments built in registers only.
// ---------------------------------------------------------------------------
__global__ __launch_bounds__(512) void sliding_attn(const __bf16* __restrict__ Qh,
                                                    const __bf16* __restrict__ Kh,
                                                    const __bf16* __restrict__ Vt,
                                                    const int* __restrict__ attend,
                                                    __bf16* __restrict__ Out) {
  const int chunk = blockIdx.x;
  const int head = blockIdx.y;
  const int b = blockIdx.z;
  const int lane = threadIdx.x & 31;
  const int wave = threadIdx.x >> 5;   // 0..15
  const int mr = lane & 15;
  const int hl = lane >> 4;
  const int mh = hl * 8;

  const size_t bh = (size_t)b * NHEADS + head;
  const __bf16* Qb = Qh + bh * SEQLEN * DHEAD;
  const __bf16* Kb = Kh + bh * SEQLEN * DHEAD;
  const __bf16* Vb = Vt + bh * DHEAD * SEQLEN;

  __shared__ unsigned char s_att[3 * CHUNK];
  __shared__ __bf16 s_p[16][16 * 32];  // per-wave P staging (C-layout -> A-layout)

  const int win0 = chunk * CHUNK - CHUNK;  // global key index of window pos 0
  for (int w = threadIdx.x; w < 3 * CHUNK; w += 512) {
    int gk = win0 + w;
    s_att[w] = (gk >= 0 && gk < SEQLEN && attend[(size_t)b * SEQLEN + gk]) ? 1 : 0;
  }
  __syncthreads();

  const int qt0 = wave * 16;  // query tile base (within chunk) for this wave
  const __bf16* qp = Qb + (size_t)(chunk * CHUNK + qt0 + mr) * DHEAD + hl * 16;

  f32x8 o[4];
  float mrun[8], lrun[8];
#pragma unroll
  for (int t = 0; t < 4; ++t)
#pragma unroll
    for (int i = 0; i < 8; ++i) o[t][i] = 0.0f;
#pragma unroll
  for (int r = 0; r < 8; ++r) {
    mrun[r] = -INFINITY;
    lrun[r] = 0.0f;
  }

  for (int jb = 0; jb < 24; ++jb) {
    const int kw0 = jb * 32;  // window offset of this 32-key block

    // ---- scores: sc[u] = Q(16x64) * K^T for the two 16-key column tiles ----
    f32x8 sc[2];
#pragma unroll
    for (int u = 0; u < 2; ++u)
#pragma unroll
      for (int i = 0; i < 8; ++i) sc[u][i] = 0.0f;

#pragma unroll
    for (int c = 0; c < 2; ++c) {  // Dh halves (K = c*32 .. c*32+31)
      bf16x16 a = load_frag2(qp + c * 32, qp + c * 32 + 8);
#pragma unroll
      for (int u = 0; u < 2; ++u) {
        // K B-fragment: lane col = key, elems i <-> d = c*32 + hl*16 + i
        int key = win0 + kw0 + u * 16 + mr;
        bool kv = (key >= 0) && (key < SEQLEN);
        const __bf16* kp = Kb + (size_t)(kv ? key : 0) * DHEAD + c * 32 + hl * 16;
        uint4 lo = *reinterpret_cast<const uint4*>(kp);
        uint4 hi = *reinterpret_cast<const uint4*>(kp + 8);
        bf16x16 f = frag_from(sel4(kv, lo), sel4(kv, hi));
        sc[u] = wmma_bf16(a, f, sc[u]);
      }
    }

    // ---- mask + scale (1/sqrt(64) = 0.125). valid: att && q <= w <= q+2C ----
#pragma unroll
    for (int u = 0; u < 2; ++u) {
      const int w = kw0 + u * 16 + mr;
      const bool att = s_att[w] != 0;
#pragma unroll
      for (int r = 0; r < 8; ++r) {
        const int q = qt0 + mh + r;
        const bool valid = att && (w >= q) && (w <= q + 2 * CHUNK);
        sc[u][r] = valid ? sc[u][r] * 0.125f : NEGV;
      }
    }

    // ---- streaming softmax across the 32 keys of this block ----
#pragma unroll
    for (int r = 0; r < 8; ++r) {
      float mx = half_max16(fmaxf(sc[0][r], sc[1][r]));
      float mnew = fmaxf(mrun[r], mx);
      float alpha = __expf(mrun[r] - mnew);
      float p0 = __expf(sc[0][r] - mnew);
      float p1 = __expf(sc[1][r] - mnew);
      float rs = half_sum16(p0 + p1);
      lrun[r] = lrun[r] * alpha + rs;
      mrun[r] = mnew;
#pragma unroll
      for (int t = 0; t < 4; ++t) o[t][r] *= alpha;
      sc[0][r] = p0;
      sc[1][r] = p1;
    }

    // ---- stage P (16x32 bf16) to per-wave LDS; reload in A-frag layout ----
#pragma unroll
    for (int u = 0; u < 2; ++u)
#pragma unroll
      for (int r = 0; r < 8; ++r)
        s_p[wave][(mh + r) * 32 + u * 16 + mr] = (__bf16)sc[u][r];

    __builtin_amdgcn_wave_barrier();  // DS ops are in-order per wave

    const __bf16* pp = &s_p[wave][mr * 32 + hl * 16];
    bf16x16 pa = load_frag2(pp, pp + 8);

    // ---- O += P * V ; V B-fragment loaded right before its WMMA ----
    const int kg0 = win0 + kw0 + hl * 16;  // keys come in 8-aligned groups
    const bool gA = (kg0 >= 0) && (kg0 < SEQLEN);
    const bool gB = (kg0 + 8 >= 0) && (kg0 + 8 < SEQLEN);
    const int kA = gA ? kg0 : 0;
    const int kB = gB ? (kg0 + 8) : 0;
#pragma unroll
    for (int t = 0; t < 4; ++t) {
      const __bf16* vrow = Vb + (size_t)(t * 16 + mr) * SEQLEN;
      uint4 r0 = *reinterpret_cast<const uint4*>(vrow + kA);
      uint4 r1 = *reinterpret_cast<const uint4*>(vrow + kB);
      bf16x16 vfr = frag_from(sel4(gA, r0), sel4(gB, r1));
      o[t] = wmma_bf16(pa, vfr, o[t]);
    }

    __builtin_amdgcn_wave_barrier();
  }

  // normalize + write to [B*S, DMODEL] bf16 at col = head*64 + d
#pragma unroll
  for (int r = 0; r < 8; ++r) {
    const float inv = 1.0f / lrun[r];
    const int sg = chunk * CHUNK + qt0 + mh + r;
    __bf16* orow = Out + ((size_t)b * SEQLEN + sg) * DMODEL + head * DHEAD;
#pragma unroll
    for (int t = 0; t < 4; ++t) orow[t * 16 + mr] = (__bf16)(o[t][r] * inv);
  }
}

// ---------------------------------------------------------------------------
// Host orchestration
// ---------------------------------------------------------------------------
extern "C" void kernel_launch(void* const* d_in, const int* in_sizes, int n_in,
                              void* d_out, int out_size, void* d_ws,
                              size_t ws_size, hipStream_t stream) {
  (void)in_sizes; (void)n_in; (void)out_size; (void)ws_size;

  const float* word_emb = (const float*)d_in[0];
  const float* pos_emb = (const float*)d_in[1];
  const float* emb_g = (const float*)d_in[2];
  const float* emb_b = (const float*)d_in[3];
  const float* Wq = (const float*)d_in[4];
  const float* bq = (const float*)d_in[5];
  const float* Wk = (const float*)d_in[6];
  const float* bk = (const float*)d_in[7];
  const float* Wv = (const float*)d_in[8];
  const float* bv = (const float*)d_in[9];
  const float* Wo = (const float*)d_in[10];
  const float* bo = (const float*)d_in[11];
  const float* ln1g = (const float*)d_in[12];
  const float* ln1b = (const float*)d_in[13];
  const float* W1 = (const float*)d_in[14];
  const float* b1 = (const float*)d_in[15];
  const float* W2 = (const float*)d_in[16];
  const float* b2 = (const float*)d_in[17];
  const float* ln2g = (const float*)d_in[18];
  const float* ln2b = (const float*)d_in[19];
  const int* tokens = (const int*)d_in[20];

  char* ws = (char*)d_ws;
  size_t off = 0;
  auto alloc = [&](size_t bytes) -> void* {
    void* p = (void*)(ws + off);
    off += (bytes + 255) & ~(size_t)255;
    return p;
  };

  float* h = (float*)alloc((size_t)MROWS * DMODEL * 4);
  __bf16* hb = (__bf16*)alloc((size_t)MROWS * DMODEL * 2);
  int* attend = (int*)alloc((size_t)MROWS * 4);
  float* qf = (float*)alloc((size_t)MROWS * DMODEL * 4);  // also Wo-out
  float* kf = (float*)alloc((size_t)MROWS * DMODEL * 4);  // also W2-out
  float* vf = (float*)alloc((size_t)MROWS * DMODEL * 4);
  __bf16* qh = (__bf16*)alloc((size_t)MROWS * DMODEL * 2);
  __bf16* kh = (__bf16*)alloc((size_t)MROWS * DMODEL * 2);
  __bf16* vt = (__bf16*)alloc((size_t)MROWS * DMODEL * 2);
  __bf16* ab = (__bf16*)alloc((size_t)MROWS * DMODEL * 2);
  __bf16* fb = (__bf16*)alloc((size_t)MROWS * DFF * 2);
  __bf16* wqt = (__bf16*)alloc((size_t)DMODEL * DMODEL * 2);
  __bf16* wkt = (__bf16*)alloc((size_t)DMODEL * DMODEL * 2);
  __bf16* wvt = (__bf16*)alloc((size_t)DMODEL * DMODEL * 2);
  __bf16* wot = (__bf16*)alloc((size_t)DMODEL * DMODEL * 2);
  __bf16* w1t = (__bf16*)alloc((size_t)DMODEL * DFF * 2);
  __bf16* w2t = (__bf16*)alloc((size_t)DFF * DMODEL * 2);

  embed_ln<<<dim3(SEQLEN, NBATCH), 256, 0, stream>>>(tokens, word_emb, pos_emb,
                                                     emb_g, emb_b, h, hb, attend);

  const int nWsq = DMODEL * DMODEL;
  const int nWff = DMODEL * DFF;
  const dim3 gP(DMODEL / 64, MROWS / 128);   // 768-wide GEMMs
  const dim3 gF(DFF / 64, MROWS / 128);      // 3072-wide GEMM
  const dim3 gAttn(NCHUNK, NHEADS, NBATCH);

  for (int l = 0; l < NLAYER; ++l) {
    const size_t wsq = (size_t)l * DMODEL * DMODEL;
    const size_t wff = (size_t)l * DMODEL * DFF;
    const size_t bm = (size_t)l * DMODEL;
    const size_t bf_ = (size_t)l * DFF;

    convert_wT<<<(nWsq + 255) / 256, 256, 0, stream>>>(Wq + wsq, wqt, DMODEL, DMODEL);
    convert_wT<<<(nWsq + 255) / 256, 256, 0, stream>>>(Wk + wsq, wkt, DMODEL, DMODEL);
    convert_wT<<<(nWsq + 255) / 256, 256, 0, stream>>>(Wv + wsq, wvt, DMODEL, DMODEL);
    convert_wT<<<(nWsq + 255) / 256, 256, 0, stream>>>(Wo + wsq, wot, DMODEL, DMODEL);
    convert_wT<<<(nWff + 255) / 256, 256, 0, stream>>>(W1 + wff, w1t, DMODEL, DFF);
    convert_wT<<<(nWff + 255) / 256, 256, 0, stream>>>(W2 + wff, w2t, DFF, DMODEL);

    gemm_wmma<0, true, false><<<gP, 256, 0, stream>>>(hb, wqt, bq + bm, qf, (__bf16*)nullptr, MROWS, DMODEL, DMODEL);
    gemm_wmma<0, true, false><<<gP, 256, 0, stream>>>(hb, wkt, bk + bm, kf, (__bf16*)nullptr, MROWS, DMODEL, DMODEL);
    gemm_wmma<0, true, false><<<gP, 256, 0, stream>>>(hb, wvt, bv + bm, vf, (__bf16*)nullptr, MROWS, DMODEL, DMODEL);

    pack_qkv<<<((size_t)MROWS * DMODEL + 255) / 256, 256, 0, stream>>>(qf, kf, vf, qh, kh, vt);

    sliding_attn<<<gAttn, 512, 0, stream>>>(qh, kh, vt, attend, ab);

    gemm_wmma<0, true, false><<<gP, 256, 0, stream>>>(ab, wot, bo + bm, qf, (__bf16*)nullptr, MROWS, DMODEL, DMODEL);
    add_ln<<<MROWS, 256, 0, stream>>>(h, qf, ln1g + bm, ln1b + bm, h, hb);

    gemm_wmma<1, false, true><<<gF, 256, 0, stream>>>(hb, w1t, b1 + bf_, (float*)nullptr, fb, MROWS, DFF, DMODEL);
    gemm_wmma<0, true, false><<<gP, 256, 0, stream>>>(fb, w2t, b2 + bm, kf, (__bf16*)nullptr, MROWS, DMODEL, DFF);
    add_ln<<<MROWS, 256, 0, stream>>>(h, kf, ln2g + bm, ln2b + bm, h, hb);
  }

  hipMemcpyAsync(d_out, h, (size_t)MROWS * DMODEL * sizeof(float),
                 hipMemcpyDeviceToDevice, stream);
}
